// GRRN_16466904613783
// MI455X (gfx1250) — matrix operand order repaired and checked by reading, not compiled, not run on hardware
//
#include <hip/hip_runtime.h>
#include <hip/hip_bf16.h>
#include <math.h>
#include <stddef.h>

typedef __attribute__((ext_vector_type(16))) _Float16 v16h;
typedef __attribute__((ext_vector_type(8)))  float    v8f;

#define BB 32
#define NN 128
#define DD 64
#define EE 5
#define TT 6   // 1 + T iterations

#if __has_builtin(__builtin_amdgcn_tensor_load_to_lds) && \
    __has_builtin(__builtin_amdgcn_s_wait_tensorcnt)
#define USE_TDM 1
#else
#define USE_TDM 0
#endif

// ---------------- LDS layout (~302 KB, fits 320 KB WGP LDS) ----------------
// All WMMA B operands live in "fragment layout": frag(kt, ct) is 32 lanes x 16
// contiguous halves, so a B fragment load is two ds_load_b128 per lane.
//   element (k, n):  lane = ((k>>4)&1)*16 + (n&15), elem = k&15,
//   addr = ((kt*4 + ct)*32 + lane)*16 + elem,  kt = k>>5, ct = n>>4.
struct SMem {
  float prop[NN * DD];        // f32 propagated state (TDM target, keep first)
  float zgate[NN * DD];       // f32 z gate
  float rowS[NN * 8];         // row attention scores, E padded to 8
  float colS[NN * 8];
  float waRow[EE * DD];
  float waCol[EE * DD];
  float beS[EE * DD];
  float baS[8];
  float brS[DD];
  float bzS[DD];
  float bhS[DD];
  unsigned maskBits[NN * 4];                  // bit-packed (N x N) mask
  alignas(32) _Float16 cat[NN * 128];         // cols 0..63 merged, 64..127 prop / r*prop
  alignas(32) _Float16 msgF[20 * 4 * 32 * 16];// msg in B-fragment layout (K=640)
  alignas(32) _Float16 rpH[NN * DD];          // r * prop (f16)
  alignas(32) _Float16 weF[EE * 2 * 4 * 32 * 16]; // We[e] fragments (K=64)
  alignas(32) _Float16 wrF[4 * 4 * 32 * 16];      // Wr fragments (K=128)
  alignas(32) _Float16 wzF[4 * 4 * 32 * 16];
  alignas(32) _Float16 whF[4 * 4 * 32 * 16];
};

__device__ __forceinline__ float sigf(float x) { return 1.0f / (1.0f + __expf(-x)); }

__device__ __forceinline__ v8f wmma_f16(v16h a, v16h b, v8f c) {
  // 8 args: (neg_a, A, neg_b, B, c_mod, C, reuse_a, reuse_b)
  return __builtin_amdgcn_wmma_f32_16x16x32_f16(false, a, false, b, (short)0, c,
                                                false, false);
}

// A fragment (16x32 f16) from row-major LDS, leading dim = ld.
__device__ __forceinline__ v16h load_a(const _Float16* p, int row0, int k0, int ld,
                                       int lane) {
  const int l = lane & 15, g = (lane >> 4) & 1;
  const _Float16* r = p + (row0 + l) * ld + k0;
  v16h a;
#pragma unroll
  for (int i = 0; i < 8; ++i) a[i] = r[g * 16 + i];
#pragma unroll
  for (int i = 0; i < 8; ++i) a[8 + i] = r[g * 16 + 8 + i];
  return a;
}

// B fragment (32x16 f16) in pre-swizzled fragment layout: contiguous 32B/lane.
__device__ __forceinline__ v16h load_b_frag(const _Float16* p, int kt, int ct,
                                            int lane) {
  return *(const v16h*)(p + ((kt * 4 + ct) * 32 + lane) * 16);
}

// Stage a (K x 64) row-major f32 matrix into fragment-layout f16.
__device__ __forceinline__ void stage_w_frag(const float* W, _Float16* dst, int K,
                                             int tid) {
  for (int idx = tid; idx < K * DD; idx += 256) {
    const int k = idx >> 6, n = idx & 63;
    const int kt = k >> 5, g = (k >> 4) & 1, i = k & 15;
    const int ct = n >> 4, l = n & 15;
    dst[((kt * 4 + ct) * 32 + (g * 16 + l)) * 16 + i] = (_Float16)W[idx];
  }
}

// On-the-fly A fragment of the score matrix S (N x N*E), K index = j*E + e.
__device__ __forceinline__ v16h load_a_scores(const SMem& S, int i0, int k0,
                                              int lane) {
  const int l = lane & 15, g = (lane >> 4) & 1;
  const int i = i0 + l;
  v16h a;
#pragma unroll
  for (int t = 0; t < 16; ++t) {
    const int koff = g * 16 + t;
    const int k = k0 + koff;
    const int j = k / EE;
    const int e = k - j * EE;
    const unsigned mbit = (S.maskBits[i * 4 + (j >> 5)] >> (j & 31)) & 1u;
    float s = 0.0f;
    if (mbit) s = sigf(S.rowS[i * 8 + e] + S.colS[j * 8 + e] + S.baS[e]);
    a[t] = (_Float16)s;
  }
  return a;
}

// One 16x16 tile of cat(128x128) @ W(128x64), W in fragment layout.
__device__ __forceinline__ v8f mm_cat(const SMem& S, const _Float16* Wf, int i0,
                                      int ct, int lane) {
  v8f acc = {};
#pragma unroll
  for (int ks = 0; ks < 4; ++ks) {
    v16h a = load_a(S.cat, i0, ks * 32, 128, lane);
    v16h b = load_b_frag(Wf, ks, ct, lane);
    acc = wmma_f16(a, b, acc);
  }
  return acc;
}

#if USE_TDM
typedef __attribute__((ext_vector_type(4))) unsigned int u32x4;
typedef __attribute__((ext_vector_type(8))) int i32x8;
typedef __attribute__((ext_vector_type(4))) int i32x4;

// 1-D TDM copy: nelem4 4-byte elements global -> LDS (one descriptor).
__device__ __forceinline__ void tdm_copy_1d(const void* gsrc, unsigned lds_off,
                                            unsigned nelem4) {
  const unsigned long long ga = (unsigned long long)(uintptr_t)gsrc;
  u32x4 g0;
  g0[0] = 1u;                                       // count=1, user descriptor
  g0[1] = lds_off;                                  // lds_addr
  g0[2] = (unsigned)(ga & 0xFFFFFFFFu);             // global_addr[31:0]
  g0[3] = (unsigned)((ga >> 32) & 0x01FFFFFFu)      // global_addr[56:32]
          | (2u << 30);                             // type = 2 ("image")
  i32x8 g1;
  g1[0] = (int)(2u << 16);                          // data_size = 4B
  g1[1] = (int)((nelem4 & 0xFFFFu) << 16);          // tensor_dim0[15:0]
  g1[2] = (int)((nelem4 >> 16) & 0xFFFFu) | (1 << 16); // dim0 hi | tensor_dim1=1
  g1[3] = (int)((nelem4 & 0xFFFFu) << 16);          // tile_dim0 (<=65535)
  g1[4] = 1;                                        // tile_dim1=1, tile_dim2=0
  g1[5] = (int)nelem4;                              // tensor_dim0_stride lo
  g1[6] = 0;                                        // stride hi / dim1_stride
  g1[7] = 0;
  const i32x4 z4 = {0, 0, 0, 0};
  const i32x8 z8 = {0, 0, 0, 0, 0, 0, 0, 0};
  // 6-arg toolchain form: (g0, g1, g2, g3, g4, cpol)
  __builtin_amdgcn_tensor_load_to_lds(g0, g1, z4, z4, z8, 0);
}
#endif

extern "C" __global__ __launch_bounds__(256, 1)
void grrn_kernel(const float* __restrict__ inp, const int* __restrict__ mask,
                 const float* __restrict__ We, const float* __restrict__ be,
                 const float* __restrict__ Wa, const float* __restrict__ ba,
                 const float* __restrict__ Wr, const float* __restrict__ br,
                 const float* __restrict__ Wz, const float* __restrict__ bz,
                 const float* __restrict__ Wh, const float* __restrict__ bh,
                 float* __restrict__ out) {
  extern __shared__ char smem_raw[];
  SMem& S = *reinterpret_cast<SMem*>(smem_raw);
  const int tid = threadIdx.x;
  const int lane = tid & 31;   // wave32
  const int wave = tid >> 5;   // 8 waves / block
  const int blk = blockIdx.x;  // one block per batch element

#if USE_TDM
  // Kick off async TDM staging of prop (f32) and the mask (into msgF scratch).
  if (wave == 0) {
    tdm_copy_1d(inp + (size_t)blk * NN * DD,
                (unsigned)offsetof(SMem, prop), NN * DD);
    tdm_copy_1d(mask + (size_t)blk * NN * NN,
                (unsigned)offsetof(SMem, msgF), NN * NN);
  }
#endif
  __builtin_prefetch(We, 0, 1);
  __builtin_prefetch(Wr, 0, 1);

  // ---- stage weights into LDS in WMMA B-fragment layout ----
  for (int e = 0; e < EE; ++e)
    stage_w_frag(We + e * DD * DD, S.weF + e * (2 * 4 * 32 * 16), DD, tid);
  stage_w_frag(Wr, S.wrF, 2 * DD, tid);
  stage_w_frag(Wz, S.wzF, 2 * DD, tid);
  stage_w_frag(Wh, S.whF, 2 * DD, tid);
  for (int p = tid; p < EE * DD; p += 256) {
    const int e = p / DD, d = p - e * DD;
    S.waRow[p] = Wa[e * 2 * DD + d];
    S.waCol[p] = Wa[e * 2 * DD + DD + d];
    S.beS[p] = be[p];
  }
  if (tid < 8) S.baS[tid] = (tid < EE) ? ba[tid] : 0.0f;
  if (tid < DD) { S.brS[tid] = br[tid]; S.bzS[tid] = bz[tid]; S.bhS[tid] = bh[tid]; }

#if USE_TDM
  if (wave == 0) __builtin_amdgcn_s_wait_tensorcnt(0);
  __syncthreads();
  // f16 copy of prop into cat cols 64..127 (prop already in LDS via TDM)
  for (int p = tid; p < NN * DD; p += 256) {
    const int i = p >> 6, d = p & 63;
    S.cat[i * 128 + 64 + d] = (_Float16)S.prop[p];
  }
  // bit-pack the mask from its LDS staging area
  {
    const unsigned* mstage = reinterpret_cast<const unsigned*>(S.msgF);
    for (int w = tid; w < NN * 4; w += 256) {
      unsigned bits = 0;
      const unsigned* mrow = mstage + w * 32;
#pragma unroll
      for (int b2 = 0; b2 < 32; ++b2) bits |= (mrow[b2] ? 1u : 0u) << b2;
      S.maskBits[w] = bits;
    }
  }
#else
  for (int p = tid; p < NN * DD; p += 256) {
    const float v = inp[blk * NN * DD + p];
    S.prop[p] = v;
    const int i = p >> 6, d = p & 63;
    S.cat[i * 128 + 64 + d] = (_Float16)v;
  }
  for (int w = tid; w < NN * 4; w += 256) {
    const int i = w >> 2, jw = w & 3;
    unsigned bits = 0;
    const int* mrow = mask + (blk * NN + i) * NN + jw * 32;
#pragma unroll
    for (int b2 = 0; b2 < 32; ++b2) bits |= (mrow[b2] ? 1u : 0u) << b2;
    S.maskBits[w] = bits;
  }
#endif
  __syncthreads();

  for (int t = 0; t < TT; ++t) {
    // zero row/col accumulators
    for (int p = tid; p < NN * 8; p += 256) { S.rowS[p] = 0.0f; S.colS[p] = 0.0f; }
    __syncthreads();

    // ---- GEMM 1: msg[j,e,:] = prop @ We[e] + be ; fused row/col reductions;
    //      msg written directly in B-fragment layout for GEMM 2 ----
    for (int tt = wave; tt < 8 * EE * 4; tt += 8) {
      const int rt = tt / (EE * 4);
      const int c2 = tt - rt * (EE * 4);
      const int e = c2 >> 2, ct = c2 & 3;
      const int i0 = rt * 16, d0 = ct * 16;
      v8f acc = {};
#pragma unroll
      for (int ks = 0; ks < 2; ++ks) {
        v16h a = load_a(S.cat + 64, i0, ks * 32, 128, lane);  // prop f16
        v16h b = load_b_frag(S.weF + e * (2 * 4 * 32 * 16), ks, ct, lane);
        acc = wmma_f16(a, b, acc);
      }
      const int l = lane & 15, g = (lane >> 4) & 1;
      const int n = d0 + l;
      const float bev = S.beS[e * DD + n];
      const float wrv = S.waRow[e * DD + n];
      const float wcv = S.waCol[e * DD + n];
#pragma unroll
      for (int r = 0; r < 8; ++r) {
        const int m = i0 + g * 8 + r;
        const float v = acc[r] + bev;
        const int k = m * EE + e;                 // merged-GEMM K index
        const int kt = k >> 5, g2 = (k >> 4) & 1, ki = k & 15;
        S.msgF[((kt * 4 + ct) * 32 + (g2 * 16 + l)) * 16 + ki] = (_Float16)v;
        atomicAdd(&S.rowS[m * 8 + e], v * wrv);   // ds_add_f32
        atomicAdd(&S.colS[m * 8 + e], v * wcv);
      }
    }
    __syncthreads();

    // ---- stream scores to global output (t, b, i, j, e) ----
    float* outT = out + ((size_t)t * BB + blk) * (size_t)(NN * NN * EE);
    for (int p = tid; p < NN * NN; p += 256) {
      const int i = p >> 7, j = p & 127;
      const unsigned mbit = (S.maskBits[i * 4 + (j >> 5)] >> (j & 31)) & 1u;
      float* o = outT + (size_t)p * EE;
#pragma unroll
      for (int e = 0; e < EE; ++e) {
        float s = 0.0f;
        if (mbit) s = sigf(S.rowS[i * 8 + e] + S.colS[j * 8 + e] + S.baS[e]);
        o[e] = s;
      }
    }

    // ---- GEMM 2: merged = S(128 x 640) @ msg(640 x 64) -> cat[:,0:64] ----
    for (int tt = wave; tt < 32; tt += 8) {
      const int i0 = (tt >> 2) * 16, ct = tt & 3, d0 = ct * 16;
      v8f acc = {};
      for (int ks = 0; ks < 20; ++ks) {
        v16h a = load_a_scores(S, i0, ks * 32, lane);  // scores built in regs
        v16h b = load_b_frag(S.msgF, ks, ct, lane);
        acc = wmma_f16(a, b, acc);
      }
      const int l = lane & 15, g = (lane >> 4) & 1;
#pragma unroll
      for (int r = 0; r < 8; ++r)
        S.cat[(i0 + g * 8 + r) * 128 + d0 + l] = (_Float16)acc[r];
    }
    __syncthreads();

    // ---- GRU gates r, z : cat @ Wr / Wz ----
    for (int tt = wave; tt < 32; tt += 8) {
      const int i0 = (tt >> 2) * 16, ct = tt & 3, d0 = ct * 16;
      const int l = lane & 15, g = (lane >> 4) & 1;
      const int n = d0 + l;
      v8f ar = mm_cat(S, S.wrF, i0, ct, lane);
      v8f az = mm_cat(S, S.wzF, i0, ct, lane);
      const float brv = S.brS[n], bzv = S.bzS[n];
#pragma unroll
      for (int r = 0; r < 8; ++r) {
        const int m = i0 + g * 8 + r;
        const float rv = sigf(ar[r] + brv);
        S.rpH[m * DD + n] = (_Float16)(rv * S.prop[m * DD + n]);
        S.zgate[m * DD + n] = sigf(az[r] + bzv);
      }
    }
    __syncthreads();

    // cat[:,64:] <- r * prop
    for (int p = tid; p < NN * DD; p += 256) {
      const int i = p >> 6, d = p & 63;
      S.cat[i * 128 + 64 + d] = S.rpH[p];
    }
    __syncthreads();

    // ---- candidate h_hat = tanh(cat @ Wh + bh); accumulate fully in regs ----
    v8f ah[4];
#pragma unroll
    for (int q = 0; q < 4; ++q) {
      const int tt = wave + q * 8;
      ah[q] = mm_cat(S, S.whF, (tt >> 2) * 16, tt & 3, lane);
    }
    __syncthreads();  // all reads of cat done before rewriting cols 64..

    // ---- prop = (1-z)*prop + z*h_hat ; refresh f16 copy ----
#pragma unroll
    for (int q = 0; q < 4; ++q) {
      const int tt = wave + q * 8;
      const int i0 = (tt >> 2) * 16, d0 = (tt & 3) * 16;
      const int l = lane & 15, g = (lane >> 4) & 1;
      const int n = d0 + l;
      const float bhv = S.bhS[n];
#pragma unroll
      for (int r = 0; r < 8; ++r) {
        const int m = i0 + g * 8 + r;
        const float h = tanhf(ah[q][r] + bhv);
        const float z = S.zgate[m * DD + n];
        const float pn = (1.0f - z) * S.prop[m * DD + n] + z * h;
        S.prop[m * DD + n] = pn;
        S.cat[m * 128 + 64 + n] = (_Float16)pn;
      }
    }
    __syncthreads();
  }
}

extern "C" void kernel_launch(void* const* d_in, const int* in_sizes, int n_in,
                              void* d_out, int out_size, void* d_ws, size_t ws_size,
                              hipStream_t stream) {
  (void)in_sizes; (void)n_in; (void)out_size; (void)d_ws; (void)ws_size;
  const float* inp  = (const float*)d_in[0];
  const int*   mask = (const int*)d_in[1];
  const float* We   = (const float*)d_in[2];
  const float* be   = (const float*)d_in[3];
  const float* Wa   = (const float*)d_in[4];
  const float* ba   = (const float*)d_in[5];
  const float* Wr   = (const float*)d_in[6];
  const float* br   = (const float*)d_in[7];
  const float* Wz   = (const float*)d_in[8];
  const float* bz   = (const float*)d_in[9];
  const float* Wh   = (const float*)d_in[10];
  const float* bh   = (const float*)d_in[11];
  float* out = (float*)d_out;

  // Idempotent, capture-safe host call: allow full-WGP dynamic LDS.
  (void)hipFuncSetAttribute((const void*)grrn_kernel,
                            hipFuncAttributeMaxDynamicSharedMemorySize,
                            (int)sizeof(SMem));
  grrn_kernel<<<BB, 256, sizeof(SMem), stream>>>(inp, mask, We, be, Wa, ba, Wr, br,
                                                 Wz, bz, Wh, bh, out);
}
